// MyLSTM_38268158608238
// MI455X (gfx1250) — matrix operand order repaired and checked by reading, not compiled, not run on hardware
//
#include <hip/hip_runtime.h>
#include <hip/hip_bf16.h>

// ---------------------------------------------------------------------------
// LSTM on gfx1250 (MI455X), wave32 + v_wmma_f32_16x16x32_bf16.
//
// Phase 0: convert/pack x and weights to bf16 (WMMA-friendly [N,K] row-major),
//          fold biases, zero h-state + barrier counters (every launch).
// Phase 1: Z[s][b][4H] = x[b,s,:] @ Wx.T + bias     (parallel WMMA GEMM)
// Phase 2: persistent 16-WG recurrence: gates = Z + h @ Wh.T, elementwise,
//          h broadcast via L2 + device-scope barrier per step.
// Phase 3: y[b,s,:] = h[b,s,:] @ Wh2o.T + b_h2o     (parallel WMMA GEMM)
//
// Workspace (~340 MB): x_bf16 32MB | Wx 2MB | Wh 2MB | Who 0.5MB | bias 8KB
//                      Z 256MB | hbuf 64KB | Hall 32MB | counters 2KB
// ---------------------------------------------------------------------------

typedef __attribute__((ext_vector_type(4)))  __bf16 v4bf;
typedef __attribute__((ext_vector_type(8)))  __bf16 v8bf;
typedef __attribute__((ext_vector_type(16))) __bf16 v16bf;
typedef __attribute__((ext_vector_type(8)))  float  v8f;

#define LSTM_B   64
#define LSTM_S   512
#define LSTM_H   512
#define LSTM_G4  2048   // 4 gates * H
#define NWG      16     // workgroups in recurrent phase
#define HSLICE   32     // hidden units per workgroup

// ---- WMMA fragment loaders (contiguous k = 16*half + s, same for A and B) -----------------

// A 16x32 bf16: lane holds row (lane&15); 16 contiguous K at k0 + half*16.
__device__ __forceinline__ v16bf frag_a(const __bf16* __restrict__ base,
                                        int ld, int row0, int k0, int lane) {
  const int m = lane & 15, half = lane >> 4;
  const __bf16* p = base + (size_t)(row0 + m) * ld + k0;
  return *(const v16bf*)(p + half * 16);
}

// B 32x16 bf16 from W stored row-major [N,K] (B[k][n] = W[n][k]):
// lane holds column n0+(lane&15); 16 contiguous K at k0 + half*16.
__device__ __forceinline__ v16bf frag_b(const __bf16* __restrict__ w,
                                        int ld, int n0, int k0, int lane) {
  const int n = lane & 15, half = lane >> 4;
  return *(const v16bf*)(w + (size_t)(n0 + n) * ld + k0 + half * 16);
}

__device__ __forceinline__ v8f wmma_bf16(v16bf a, v16bf b, v8f c) {
  return __builtin_amdgcn_wmma_f32_16x16x32_bf16(false, a, false, b,
                                                 (short)0, c, false, false);
}

__device__ __forceinline__ float sigmoid_f(float x) {
  return 1.0f / (1.0f + __expf(-x));
}

// ---- Phase 0a: x (fp32) -> bf16, vectorized ------------------------------
__global__ __launch_bounds__(256) void k_conv_x(const float4* __restrict__ x,
                                                v4bf* __restrict__ xb, int n4) {
  int stride = gridDim.x * blockDim.x;
  for (int i = blockIdx.x * blockDim.x + threadIdx.x; i < n4; i += stride) {
    float4 v = x[i];
    v4bf o;
    o[0] = (__bf16)v.x; o[1] = (__bf16)v.y; o[2] = (__bf16)v.z; o[3] = (__bf16)v.w;
    xb[i] = o;
  }
}

// ---- Phase 0b: pack weights/biases, zero h + barrier counters ------------
__global__ __launch_bounds__(256) void k_prep(
    const float* __restrict__ Wi, const float* __restrict__ Wf,
    const float* __restrict__ Wo, const float* __restrict__ Wc,
    const float* __restrict__ bi, const float* __restrict__ bf_,
    const float* __restrict__ bo, const float* __restrict__ bc,
    const float* __restrict__ Who,
    __bf16* __restrict__ Wx, __bf16* __restrict__ Wh,
    __bf16* __restrict__ Whop, float* __restrict__ biasp,
    __bf16* __restrict__ hbuf, unsigned int* __restrict__ cnt) {
  const int stride = gridDim.x * blockDim.x;
  const int tid0 = blockIdx.x * blockDim.x + threadIdx.x;
  // gate weight split: W_g is [H, I+H]; cols 0..511 -> Wx, 512..1023 -> Wh
  for (int i = tid0; i < 4 * LSTM_H * LSTM_H; i += stride) {
    int gate = i >> 18;               // / (512*512)
    int rem  = i & (LSTM_H * LSTM_H - 1);
    int n = rem >> 9, k = rem & 511;
    const float* Wg = (gate == 0) ? Wi : (gate == 1) ? Wf : (gate == 2) ? Wo : Wc;
    size_t dst = ((size_t)gate * LSTM_H + n) * LSTM_H + k;
    Wx[dst] = (__bf16)Wg[(size_t)n * 1024 + k];
    Wh[dst] = (__bf16)Wg[(size_t)n * 1024 + 512 + k];
  }
  for (int i = tid0; i < LSTM_H * LSTM_H; i += stride)
    Whop[i] = (__bf16)Who[i];
  for (int i = tid0; i < LSTM_G4; i += stride) {
    int gate = i >> 9, n = i & 511;
    const float* bg = (gate == 0) ? bi : (gate == 1) ? bf_ : (gate == 2) ? bo : bc;
    biasp[i] = bg[n];
  }
  for (int i = tid0; i < LSTM_B * LSTM_H; i += stride)
    hbuf[i] = (__bf16)0.0f;
  for (int i = tid0; i < LSTM_S; i += stride)
    cnt[i] = 0u;
}

// ---- Phase 1: Z[s][b][4H] = x @ Wx.T + bias ------------------------------
// block = 8 waves = 64-row x 256-col tile; grid = (2048/256, 32768/64)
__global__ __launch_bounds__(256) void k_pregemm(
    const __bf16* __restrict__ xb,   // [B*S, 512] row-major (row = b*S+s)
    const __bf16* __restrict__ Wx,   // [2048, 512]
    const float*  __restrict__ bias, // [2048]
    float* __restrict__ Z) {         // [S][B][2048]
  const int lane = threadIdx.x & 31;
  const int wv = threadIdx.x >> 5;
  const int wm = wv & 1, wn = wv >> 1;
  const int row0 = blockIdx.y * 64 + wm * 32;   // 2 m-tiles per wave
  const int col0 = blockIdx.x * 256 + wn * 64;  // 4 n-tiles per wave
  v8f acc[2][4];
#pragma unroll
  for (int i = 0; i < 2; ++i)
#pragma unroll
    for (int j = 0; j < 4; ++j)
#pragma unroll
      for (int r = 0; r < 8; ++r) acc[i][j][r] = 0.0f;

  for (int k0 = 0; k0 < LSTM_H; k0 += 32) {
    v16bf a0 = frag_a(xb, LSTM_H, row0, k0, lane);
    v16bf a1 = frag_a(xb, LSTM_H, row0 + 16, k0, lane);
#pragma unroll
    for (int j = 0; j < 4; ++j) {
      v16bf b = frag_b(Wx, LSTM_H, col0 + 16 * j, k0, lane);
      acc[0][j] = wmma_bf16(a0, b, acc[0][j]);
      acc[1][j] = wmma_bf16(a1, b, acc[1][j]);
    }
  }
  const int half = lane >> 4, nl = lane & 15;
#pragma unroll
  for (int i = 0; i < 2; ++i)
#pragma unroll
    for (int j = 0; j < 4; ++j) {
      const int n = col0 + j * 16 + nl;
      const float bv = bias[n];
#pragma unroll
      for (int r = 0; r < 8; ++r) {
        const int m = row0 + i * 16 + r + half * 8;  // m = b*S + s
        const int b = m >> 9, s = m & 511;
        Z[((size_t)s * LSTM_B + b) * LSTM_G4 + n] = acc[i][j][r] + bv;
      }
    }
}

// ---- Phase 2: persistent recurrence --------------------------------------
// 16 WGs; WG g owns hidden cols [g*32, g*32+32) => 128 gate cols.
// Wave w: m-tile (w&3), n-tiles j = 4*(w>>2)..+3 (j->gate j>>1, half (j&1)).
__global__ __launch_bounds__(256) void k_recurrent(
    const float*  __restrict__ Z,     // [S][B][2048]
    const __bf16* __restrict__ Wh,    // [2048, 512]
    __bf16* __restrict__ hbuf,        // [B, 512]  (bf16 state, L2-resident)
    __bf16* __restrict__ Hall,        // [S][B][512]
    float* __restrict__ hlast, float* __restrict__ clast,
    unsigned int* __restrict__ cnt) { // [S]
  __shared__ float gbuf[LSTM_B][132];      // gate pre-activations (pad 132)
  __shared__ float cstate[LSTM_B][HSLICE]; // cell state slice (fp32)

  const int g = blockIdx.x;
  const int tid = threadIdx.x;
  const int lane = tid & 31, wv = tid >> 5;
  const int mt = wv & 3;
  const int jbase = (wv >> 2) * 4;
  const int half = lane >> 4, nl = lane & 15;

  for (int i = tid; i < LSTM_B * HSLICE; i += 256)
    cstate[i >> 5][i & 31] = 0.0f;
  __syncthreads();

  for (int t = 0; t < LSTM_S; ++t) {
    const float* Zt = Z + (size_t)t * LSTM_B * LSTM_G4;
    v8f acc[4];
#pragma unroll
    for (int j4 = 0; j4 < 4; ++j4) {  // C init from precomputed Z (+bias)
      const int j = jbase + j4;
      const int ngl = (j >> 1) * LSTM_H + g * HSLICE + (j & 1) * 16 + nl;
#pragma unroll
      for (int r = 0; r < 8; ++r) {
        const int b = mt * 16 + r + half * 8;
        acc[j4][r] = Zt[(size_t)b * LSTM_G4 + ngl];
      }
    }
    for (int k0 = 0; k0 < LSTM_H; k0 += 32) {  // += h @ Wh.T
      v16bf a = frag_a(hbuf, LSTM_H, mt * 16, k0, lane);
#pragma unroll
      for (int j4 = 0; j4 < 4; ++j4) {
        const int j = jbase + j4;
        const int nrow = (j >> 1) * LSTM_H + g * HSLICE + (j & 1) * 16;
        v16bf b = frag_b(Wh, LSTM_H, nrow, k0, lane);
        acc[j4] = wmma_bf16(a, b, acc[j4]);
      }
    }
#pragma unroll
    for (int j4 = 0; j4 < 4; ++j4) {  // pre-activations -> LDS
      const int j = jbase + j4;
#pragma unroll
      for (int r = 0; r < 8; ++r)
        gbuf[mt * 16 + r + half * 8][j * 16 + nl] = acc[j4][r];
    }
    __syncthreads();

    // elementwise gates: 64*32 = 2048 cells, 8 per thread (fixed ownership)
#pragma unroll
    for (int e = 0; e < 8; ++e) {
      const int idx = e * 256 + tid;
      const int b = idx >> 5, hc = idx & 31;
      const float fi = sigmoid_f(gbuf[b][hc]);            // input gate
      const float ff = sigmoid_f(gbuf[b][32 + hc]);       // forget gate
      const float fo = sigmoid_f(gbuf[b][64 + hc]);       // output gate
      const float gg = tanhf(gbuf[b][96 + hc]);           // candidate
      const float c = ff * cstate[b][hc] + fi * gg;
      const float h = fo * tanhf(c);
      cstate[b][hc] = c;
      const int hcol = g * HSLICE + hc;
      const __bf16 hb = (__bf16)h;
      hbuf[(size_t)b * LSTM_H + hcol] = hb;
      Hall[((size_t)t * LSTM_B + b) * LSTM_H + hcol] = hb;
      if (t == LSTM_S - 1) {
        hlast[(size_t)b * LSTM_H + hcol] = h;
        clast[(size_t)b * LSTM_H + hcol] = c;
      }
    }

    // device-scope barrier: make h visible, then sync all 16 WGs
    __threadfence();
    __syncthreads();
    if (tid == 0) {
      __hip_atomic_fetch_add(&cnt[t], 1u, __ATOMIC_ACQ_REL,
                             __HIP_MEMORY_SCOPE_AGENT);
      while (__hip_atomic_load(&cnt[t], __ATOMIC_ACQUIRE,
                               __HIP_MEMORY_SCOPE_AGENT) < NWG) {
        __builtin_amdgcn_s_sleep(1);
      }
    }
    __syncthreads();
  }
}

// ---- Phase 3: y[b][s][:] = h[s][b][:] @ Wh2o.T + b_h2o -------------------
// grid = (512/256, 32768/64); A rows m' = s*64 + b.
__global__ __launch_bounds__(256) void k_outproj(
    const __bf16* __restrict__ Hall,  // [S*B, 512]
    const __bf16* __restrict__ Who,   // [512, 512]
    const float*  __restrict__ bho,   // [512]
    float* __restrict__ out) {        // [B][S][512]
  const int lane = threadIdx.x & 31;
  const int wv = threadIdx.x >> 5;
  const int wm = wv & 1, wn = wv >> 1;
  const int row0 = blockIdx.y * 64 + wm * 32;
  const int col0 = blockIdx.x * 256 + wn * 64;
  v8f acc[2][4];
#pragma unroll
  for (int i = 0; i < 2; ++i)
#pragma unroll
    for (int j = 0; j < 4; ++j)
#pragma unroll
      for (int r = 0; r < 8; ++r) acc[i][j][r] = 0.0f;

  for (int k0 = 0; k0 < LSTM_H; k0 += 32) {
    v16bf a0 = frag_a(Hall, LSTM_H, row0, k0, lane);
    v16bf a1 = frag_a(Hall, LSTM_H, row0 + 16, k0, lane);
#pragma unroll
    for (int j = 0; j < 4; ++j) {
      v16bf b = frag_b(Who, LSTM_H, col0 + 16 * j, k0, lane);
      acc[0][j] = wmma_bf16(a0, b, acc[0][j]);
      acc[1][j] = wmma_bf16(a1, b, acc[1][j]);
    }
  }
  const int half = lane >> 4, nl = lane & 15;
#pragma unroll
  for (int i = 0; i < 2; ++i)
#pragma unroll
    for (int j = 0; j < 4; ++j) {
      const int n = col0 + j * 16 + nl;
      const float bv = bho[n];
#pragma unroll
      for (int r = 0; r < 8; ++r) {
        const int m = row0 + i * 16 + r + half * 8;  // m = s*64 + b
        const int b = m & 63, s = m >> 6;
        out[((size_t)b * LSTM_S + s) * LSTM_H + n] = acc[i][j][r] + bv;
      }
    }
}

// ---------------------------------------------------------------------------
extern "C" void kernel_launch(void* const* d_in, const int* in_sizes, int n_in,
                              void* d_out, int out_size, void* d_ws,
                              size_t ws_size, hipStream_t stream) {
  const float* x    = (const float*)d_in[0];
  const float* W_i  = (const float*)d_in[1];
  const float* b_i  = (const float*)d_in[2];
  const float* W_f  = (const float*)d_in[3];
  const float* b_f  = (const float*)d_in[4];
  const float* W_o  = (const float*)d_in[5];
  const float* b_o  = (const float*)d_in[6];
  const float* W_c  = (const float*)d_in[7];
  const float* b_c  = (const float*)d_in[8];
  const float* W_h2o = (const float*)d_in[9];
  const float* b_h2o = (const float*)d_in[10];

  // workspace layout (256-byte aligned)
  char* ws = (char*)d_ws;
  size_t off = 0;
  auto take = [&](size_t bytes) {
    char* p = ws + off;
    off = (off + bytes + 255) & ~(size_t)255;
    return p;
  };
  __bf16* xb    = (__bf16*)take((size_t)LSTM_B * LSTM_S * LSTM_H * 2);  // 32MB
  __bf16* Wx    = (__bf16*)take((size_t)LSTM_G4 * LSTM_H * 2);          // 2MB
  __bf16* Wh    = (__bf16*)take((size_t)LSTM_G4 * LSTM_H * 2);          // 2MB
  __bf16* Whop  = (__bf16*)take((size_t)LSTM_H * LSTM_H * 2);           // 512KB
  float*  biasp = (float*)take((size_t)LSTM_G4 * 4);                    // 8KB
  float*  Z     = (float*)take((size_t)LSTM_S * LSTM_B * LSTM_G4 * 4);  // 256MB
  __bf16* hbuf  = (__bf16*)take((size_t)LSTM_B * LSTM_H * 2);           // 64KB
  __bf16* Hall  = (__bf16*)take((size_t)LSTM_S * LSTM_B * LSTM_H * 2);  // 32MB
  unsigned int* cnt = (unsigned int*)take((size_t)LSTM_S * 4);          // 2KB

  float* out   = (float*)d_out;                       // [B,S,O]
  float* hlast = out + (size_t)LSTM_B * LSTM_S * LSTM_H;
  float* clast = hlast + (size_t)LSTM_B * LSTM_H;

  // Phase 0
  const int n4 = (LSTM_B * LSTM_S * LSTM_H) / 4;
  k_conv_x<<<4096, 256, 0, stream>>>((const float4*)x, (v4bf*)xb, n4);
  k_prep<<<1024, 256, 0, stream>>>(W_i, W_f, W_o, W_c, b_i, b_f, b_o, b_c,
                                   W_h2o, Wx, Wh, Whop, biasp, hbuf, cnt);
  // Phase 1: pre-GEMM
  k_pregemm<<<dim3(LSTM_G4 / 256, (LSTM_B * LSTM_S) / 64), 256, 0, stream>>>(
      xb, Wx, biasp, Z);
  // Phase 2: recurrence (persistent, 16 cooperating WGs)
  k_recurrent<<<NWG, 256, 0, stream>>>(Z, Wh, hbuf, Hall, hlast, clast, cnt);
  // Phase 3: output projection
  k_outproj<<<dim3(LSTM_H / 256, (LSTM_B * LSTM_S) / 64), 256, 0, stream>>>(
      Hall, Whop, b_h2o, out);
}
